// InpaintGCModel_4432406249763
// MI455X (gfx1250) — matrix operand order repaired and checked by reading, not compiled, not run on hardware
//
#include <hip/hip_runtime.h>
#include <hip/hip_bf16.h>
#include <math.h>

typedef __attribute__((ext_vector_type(16))) _Float16 v16h;
typedef __attribute__((ext_vector_type(8)))  _Float16 v8h;
typedef __attribute__((ext_vector_type(8)))  float    v8f;

#define TPB 256
#define WAVES_PER_BLOCK 8
#define CONV_STRIP 16384

// ===================== Packed-fragment WMMA GEMM ====================
// A: f16 row-major [M][Kp], Kp multiple of 32, zero-padded.
// Bp: f16 fragment-packed [Kp/32][N16][2][16]:
//     lane (hf,n) of a wave reads B k-values k0+hf*8+e (e<8), k0+16+hf*8+e-8
//     (e>=8) as one contiguous 32-byte vector. N16 = N rounded up to 16, zero-padded.
// Per K-chunk the inner loop is: 2x16B A loads, 1x32B B load, 1 WMMA.
__global__ __launch_bounds__(TPB)
void k_gemm_f16(const _Float16* __restrict__ A, int Kp,
                const _Float16* __restrict__ Bp, int N16,
                const float* __restrict__ bias,
                float* __restrict__ C, int ldc, int M, int N) {
  const int mt = blockIdx.x * WAVES_PER_BLOCK + (threadIdx.x >> 5);
  if (mt * 16 >= M) return;                    // uniform per wave
  const int m0 = mt * 16, n0 = blockIdx.y * 16;
  const int lane = threadIdx.x & 31;
  const int hf = lane >> 4, mn = lane & 15;
  int mrow = m0 + mn; if (mrow >= M) mrow = M - 1;   // dup row; masked on store
  const _Float16* ap = A + (size_t)mrow * Kp + hf * 8;
  const _Float16* bp = Bp + (((size_t)n0 + mn) * 2 + hf) * 16;
  const size_t bstep = (size_t)N16 * 32;
  v8f acc = {};
  const int chunks = Kp >> 5;
  for (int ch = 0; ch < chunks; ++ch) {
    v8h a0 = *(const v8h*)ap;
    v8h a1 = *(const v8h*)(ap + 16);
    v16h a = __builtin_shufflevector(a0, a1, 0, 1, 2, 3, 4, 5, 6, 7,
                                     8, 9, 10, 11, 12, 13, 14, 15);
    v16h b = *(const v16h*)bp;
    __builtin_prefetch(ap + 32, 0, 0);
    __builtin_prefetch(bp + bstep, 0, 0);
    acc = __builtin_amdgcn_wmma_f32_16x16x32_f16(false, a, false, b,
                                                 (short)0, acc, false, false);
    ap += 32;
    bp += bstep;
  }
#pragma unroll
  for (int r = 0; r < 8; ++r) {
    const int mm = m0 + r + (hf ? 8 : 0);
    const int nn = n0 + mn;
    if (mm < M && nn < N)
      C[(size_t)mm * ldc + nn] = acc[r] + (bias ? bias[nn] : 0.f);
  }
}

// ======================= Pack kernels (one-pass) =====================
// im2col: NHWC input -> f16 A[rows][Kp]; K = (ky*KW+kx)*Cin+ci; SAME pad.
__global__ void k_im2col(const float* __restrict__ in, int H, int W, int Cin,
                         int Wo, int stride, int KH, int KW, int padH, int padW,
                         int r0, long long rows, int Kp, int Ktot,
                         _Float16* __restrict__ A) {
  long long i = blockIdx.x * (long long)blockDim.x + threadIdx.x;
  if (i >= rows * Kp) return;
  const int k = (int)(i % Kp);
  const long long rr = i / Kp;
  float v = 0.f;
  if (k < Ktot) {
    const int m = r0 + (int)rr;
    const int oh = m / Wo, ow = m - oh * Wo;
    const int ci = k % Cin;
    const int r = k / Cin;
    const int kx = r % KW, ky = r / KW;
    const int ih = oh * stride - padH + ky;
    const int iw = ow * stride - padW + kx;
    if ((unsigned)ih < (unsigned)H && (unsigned)iw < (unsigned)W)
      v = in[((size_t)ih * W + iw) * Cin + ci];
  }
  A[i] = (_Float16)v;
}

// generic f32 matrix -> f16 A[M][Kp]
__global__ void k_packA(const float* __restrict__ S, int lda, long long M, int K,
                        int Kp, _Float16* __restrict__ A) {
  long long i = blockIdx.x * (long long)blockDim.x + threadIdx.x;
  if (i >= M * Kp) return;
  const int k = (int)(i % Kp);
  const long long m = i / Kp;
  A[i] = (_Float16)((k < K) ? S[m * lda + k] : 0.f);
}

// weights (or transposed matrix) -> fragment-packed B
__global__ void k_packB(const float* __restrict__ w, int K, int N, int ldw,
                        int transB, int N16, long long total,
                        _Float16* __restrict__ Bp) {
  long long i = blockIdx.x * (long long)blockDim.x + threadIdx.x;
  if (i >= total) return;
  const int e = (int)(i & 15);
  const int hf = (int)((i >> 4) & 1);
  const long long r = i >> 5;
  const int n = (int)(r % N16);
  const int chunk = (int)(r / N16);
  const int k = chunk * 32 + (e < 8 ? hf * 8 + e : 16 + hf * 8 + (e - 8));
  float v = 0.f;
  if (k < K && n < N)
    v = transB ? w[(size_t)n * ldw + k] : w[(size_t)k * ldw + n];
  Bp[i] = (_Float16)v;
}

// 3x3 SAME patches of f (n x n x C) -> f16 A[n*n][Kp]
__global__ void k_patchA(const float* __restrict__ f, int n, int C, int Kp,
                         _Float16* __restrict__ A) {
  long long total = (long long)n * n * Kp;
  long long i = blockIdx.x * (long long)blockDim.x + threadIdx.x;
  if (i >= total) return;
  const int k = (int)(i % Kp);
  const int m = (int)(i / Kp);
  float v = 0.f;
  if (k < 9 * C) {
    const int ci = k % C, r = k / C, kx = r % 3, ky = r / 3;
    const int fy = m / n + ky - 1, fx = m % n + kx - 1;
    if ((unsigned)fy < (unsigned)n && (unsigned)fx < (unsigned)n)
      v = f[((size_t)fy * n + fx) * C + ci];
  }
  A[i] = (_Float16)v;
}

// normalized 3x3 patches of b -> fragment-packed B (column l scaled by invn[l])
__global__ void k_patchB(const float* __restrict__ b, const float* __restrict__ invn,
                         int n, int C, int N16, long long total,
                         _Float16* __restrict__ Bp) {
  long long i = blockIdx.x * (long long)blockDim.x + threadIdx.x;
  if (i >= total) return;
  const int e = (int)(i & 15);
  const int hf = (int)((i >> 4) & 1);
  const long long r = i >> 5;
  const int l = (int)(r % N16);
  const int chunk = (int)(r / N16);
  const int k = chunk * 32 + (e < 8 ? hf * 8 + e : 16 + hf * 8 + (e - 8));
  float v = 0.f;
  if (k < 9 * C && l < n * n) {
    const int ci = k % C, rr = k / C, kx = rr % 3, ky = rr / 3;
    const int by = l / n + ky - 1, bx = l % n + kx - 1;
    if ((unsigned)by < (unsigned)n && (unsigned)bx < (unsigned)n)
      v = b[((size_t)by * n + bx) * C + ci] * invn[l];
  }
  Bp[i] = (_Float16)v;
}

// ======================= Elementwise kernels ========================
__global__ void k_pack(const float* __restrict__ x, const float* __restrict__ m,
                       float* __restrict__ o, long long HW) {
  long long i = blockIdx.x * (long long)blockDim.x + threadIdx.x;
  if (i >= HW) return;
  o[i * 5 + 0] = x[i * 3 + 0];
  o[i * 5 + 1] = x[i * 3 + 1];
  o[i * 5 + 2] = x[i * 3 + 2];
  o[i * 5 + 3] = 1.f;
  o[i * 5 + 4] = m[i];
}

__global__ void k_gate(const float* __restrict__ y, float* __restrict__ o,
                       long long HW, int C, int relu) {
  long long i = blockIdx.x * (long long)blockDim.x + threadIdx.x;
  if (i >= HW * C) return;
  long long p = i / C; int c = (int)(i - p * C);
  float a = y[p * 2 * C + c];
  float g = y[p * 2 * C + C + c];
  a = relu ? fmaxf(a, 0.f) : (a > 0.f ? a : expm1f(a));
  o[i] = a * (1.f / (1.f + expf(-g)));
}

__global__ void k_up2(const float* __restrict__ in, int H, int W, int C,
                      float* __restrict__ o) {
  long long N = (long long)4 * H * W * C;
  long long i = blockIdx.x * (long long)blockDim.x + threadIdx.x;
  if (i >= N) return;
  int c = (int)(i % C); long long r = i / C;
  int x2 = (int)(r % (2 * W)); int y2 = (int)(r / (2 * W));
  o[i] = in[((size_t)(y2 >> 1) * W + (x2 >> 1)) * C + c];
}

__global__ void k_subsample(const float* __restrict__ in, int W, int C, int s,
                            int ho, int wo, float* __restrict__ o) {
  long long N = (long long)ho * wo * C;
  long long i = blockIdx.x * (long long)blockDim.x + threadIdx.x;
  if (i >= N) return;
  int c = (int)(i % C); long long r = i / C;
  int ox = (int)(r % wo); int oy = (int)(r / wo);
  o[i] = in[((size_t)(oy * s) * W + (size_t)ox * s) * C + c];
}

__global__ void k_copych(float* __restrict__ dst, int Cd, int off,
                         const float* __restrict__ src, long long HW, int Cs) {
  long long i = blockIdx.x * (long long)blockDim.x + threadIdx.x;
  if (i >= HW * Cs) return;
  long long p = i / Cs; int c = (int)(i - p * Cs);
  dst[p * Cd + off + c] = src[i];
}

__global__ void k_maxpool2(const float* __restrict__ in, int H, int W, int C,
                           float* __restrict__ o) {
  long long N = (long long)(H / 2) * (W / 2) * C;
  long long i = blockIdx.x * (long long)blockDim.x + threadIdx.x;
  if (i >= N) return;
  int c = (int)(i % C); long long r = i / C;
  int ox = (int)(r % (W / 2)); int oy = (int)(r / (W / 2));
  const float* p = in + ((size_t)(2 * oy) * W + 2 * ox) * C + c;
  o[i] = fmaxf(fmaxf(p[0], p[C]), fmaxf(p[(size_t)W * C], p[(size_t)W * C + C]));
}

__global__ void k_addsig(float* __restrict__ o, const float* __restrict__ a,
                         const float* __restrict__ b, const float* __restrict__ sig,
                         long long N) {
  long long i = blockIdx.x * (long long)blockDim.x + threadIdx.x;
  if (i >= N) return;
  o[i] = a[i] + sig[0] * b[i];
}

__global__ void k_tanh(const float* __restrict__ a, float* __restrict__ o, long long N) {
  long long i = blockIdx.x * (long long)blockDim.x + threadIdx.x;
  if (i >= N) return;
  o[i] = tanhf(a[i]);
}

__global__ void k_mm(const float* __restrict__ m, int n, float* __restrict__ mm) {
  int l = blockIdx.x * blockDim.x + threadIdx.x;
  if (l >= n * n) return;
  int ly = l / n, lx = l % n;
  float s = 0.f;
  for (int dy = -1; dy <= 1; ++dy)
    for (int dx = -1; dx <= 1; ++dx) {
      int y = ly + dy, x = lx + dx;
      if ((unsigned)y < (unsigned)n && (unsigned)x < (unsigned)n) s += m[y * n + x];
    }
  mm[l] = (s == 0.f) ? 1.f : 0.f;
}

__global__ void k_invnorm(const float* __restrict__ b, int n, int C,
                          float* __restrict__ invn) {
  int l = blockIdx.x * blockDim.x + threadIdx.x;
  if (l >= n * n) return;
  int ly = l / n, lx = l % n;
  float s = 0.f;
  for (int dy = -1; dy <= 1; ++dy)
    for (int dx = -1; dx <= 1; ++dx) {
      int y = ly + dy, x = lx + dx;
      if ((unsigned)y < (unsigned)n && (unsigned)x < (unsigned)n) {
        const float* p = b + ((size_t)y * n + x) * C;
        for (int c = 0; c < C; ++c) { float v = p[c]; s += v * v; }
      }
    }
  invn[l] = 1.f / fmaxf(sqrtf(s), 1e-4f);
}

__global__ void k_fuse1(const float* __restrict__ in, float* __restrict__ out,
                        long long M, long long L) {
  long long i = blockIdx.x * (long long)blockDim.x + threadIdx.x;
  if (i >= M * L) return;
  long long q = i / L, l = i - q * L;
  float s = 0.f;
  for (int d = -1; d <= 1; ++d) {
    long long qd = q + d, ld = l + d;
    if (qd >= 0 && qd < M && ld >= 0 && ld < L) s += in[qd * L + ld];
  }
  out[i] = s;
}

__global__ void k_fuse2(const float* __restrict__ in, float* __restrict__ out, int n) {
  long long M = (long long)n * n, L = M;
  long long i = blockIdx.x * (long long)blockDim.x + threadIdx.x;
  if (i >= M * L) return;
  long long q = i / L, l = i - q * L;
  int hi = (int)(q / n), wi = (int)(q % n);
  int hb = (int)(l / n), wb = (int)(l % n);
  long long q2 = (long long)wi * n + hi;
  long long l2 = (long long)wb * n + hb;
  float s = 0.f;
  for (int d = -1; d <= 1; ++d) {
    long long qd = q2 + d, ld = l2 + d;
    if (qd >= 0 && qd < M && ld >= 0 && ld < L) {
      int wi2 = (int)(qd / n), hi2 = (int)(qd % n);
      int wb2 = (int)(ld / n), hb2 = (int)(ld % n);
      s += in[((long long)hi2 * n + wi2) * L + (long long)hb2 * n + wb2];
    }
  }
  out[i] = s;
}

__global__ __launch_bounds__(TPB)
void k_softmax_rows(float* __restrict__ d, const float* __restrict__ cm,
                    long long L, float scale) {
  long long row = blockIdx.x;
  float* p = d + row * L;
  __shared__ float red[TPB];
  float mx = -1e30f;
  for (long long i = threadIdx.x; i < L; i += blockDim.x) {
    float v = p[i] * (cm ? cm[i] : 1.f) * scale;
    mx = fmaxf(mx, v);
  }
  red[threadIdx.x] = mx; __syncthreads();
  for (int s = TPB / 2; s > 0; s >>= 1) {
    if (threadIdx.x < (unsigned)s)
      red[threadIdx.x] = fmaxf(red[threadIdx.x], red[threadIdx.x + s]);
    __syncthreads();
  }
  mx = red[0]; __syncthreads();
  float sum = 0.f;
  for (long long i = threadIdx.x; i < L; i += blockDim.x) {
    float v = p[i] * (cm ? cm[i] : 1.f) * scale;
    float e = expf(v - mx);
    p[i] = e;
    sum += e;
  }
  red[threadIdx.x] = sum; __syncthreads();
  for (int s = TPB / 2; s > 0; s >>= 1) {
    if (threadIdx.x < (unsigned)s) red[threadIdx.x] += red[threadIdx.x + s];
    __syncthreads();
  }
  sum = red[0]; __syncthreads();
  float inv = 1.f / sum;
  for (long long i = threadIdx.x; i < L; i += blockDim.x)
    p[i] = p[i] * inv * (cm ? cm[i] : 1.f);
}

// transposed-conv gather of attention scores with 4x4 stride-2 patches of b
__global__ void k_ca_deconv(const float* __restrict__ yi,
                            const float* __restrict__ bf,
                            float* __restrict__ out, int n, int C) {
  const int S = 2 * n; const long long L = (long long)n * n;
  const int p = blockIdx.x;
  const int ny = p / S, nx = p % S;
  const int c = threadIdx.x;
  if (c >= C) return;
  const int ky0 = (ny + 1) & 1, kx0 = (nx + 1) & 1;
  float s = 0.f;
  for (int a = 0; a < 2; ++a) {
    const int ky = ky0 + 2 * a;
    const int qy = (ny + 1 - ky) >> 1;
    if ((unsigned)qy >= (unsigned)n) continue;
    for (int b2 = 0; b2 < 2; ++b2) {
      const int kx = kx0 + 2 * b2;
      const int qx = (nx + 1 - kx) >> 1;
      if ((unsigned)qx >= (unsigned)n) continue;
      const float* yr = yi + ((long long)qy * n + qx) * L;
      for (int ly = 0; ly < n; ++ly) {
        const int by = ly * 2 + ky - 1;
        if ((unsigned)by >= (unsigned)S) continue;
        const float* br = bf + ((size_t)by * S) * C + c;
        const float* yl = yr + (long long)ly * n;
        for (int lx = 0; lx < n; ++lx) {
          const int bx = lx * 2 + kx - 1;
          if ((unsigned)bx >= (unsigned)S) continue;
          s += yl[lx] * br[(size_t)bx * C];
        }
      }
    }
  }
  out[((size_t)ny * S + nx) * C + c] = s * 0.25f;
}

// ========================= Host orchestration =======================
struct Spec { int cin, cout, k; };
enum {
  S_conv1, S_conv1_ds, S_conv2, S_conv2_ds, S_conv3, S_conv3_ds, S_conv4,
  S_conv4_ds, S_conv5, S_conv5_ds, S_conv6, S_conv7, S_conv000_ds, S_conv001,
  S_conv001_ds, S_conv002, S_conv002_ds, S_conv003, S_conv7_up, S_conv8_up,
  S_conv81_up, S_conv_self32_1, S_pmconv32, S_conv9, S_conv9_up, S_conv10,
  S_conv10_up, S_conv101_up, S_conv10_1, S_conv10_2, S_pmconv64_d, S_pmconv64,
  S_conv11_d, S_conv11, S_conv11_up_d, S_conv11_up, S_conv12, S_conv12_up,
  S_conv121_up, S_conv12_1, S_conv12_2, S_pmconv128, S_pmconv128_d, S_conv13_d,
  S_conv13, S_conv13_up_d, S_conv13_up, S_conv14, S_conv14_up, S_conv15_up,
  S_conv15, S_conv16, S_conv17, NSPEC
};
static const Spec SPECS[NSPEC] = {
  {5,48,5},{24,96,3},{48,96,3},{48,192,3},{96,192,3},{96,192,3},{96,192,3},
  {96,192,3},{96,192,3},{96,192,3},{96,192,3},{96,48,3},{3,96,3},{48,96,3},
  {48,96,3},{48,96,3},{48,96,3},{48,96,3},{120,48,3},{120,48,3},
  {24,48,3},{144,96,3},{48,96,3},{48,96,3},{48,96,3},{48,96,3},
  {48,48,3},{24,48,3},{192,96,3},{96,96,3},{48,96,3},{48,96,3},
  {48,96,3},{48,96,3},{48,96,3},{48,96,3},{48,96,3},{48,48,3},
  {24,48,3},{144,96,3},{96,96,3},{48,96,3},{48,96,3},{48,96,3},
  {48,96,3},{48,96,3},{48,96,3},{48,48,3},{48,48,3},{24,48,3},
  {120,48,3},{24,24,3},{12,3,3}
};

enum { ACT_ELU = 0, ACT_RELU = 1, ACT_NONE = 2 };

struct Ctx {
  void* const* din;
  char* ws; size_t cap; size_t off;
  hipStream_t st;
};

static inline const float* PW(Ctx& c, int i) { return (const float*)c.din[3 + 2 * i]; }
static inline const float* PB(Ctx& c, int i) { return (const float*)c.din[4 + 2 * i]; }

static void* aallocB(Ctx& c, size_t bytes) {
  c.off = (c.off + 255) & ~(size_t)255;
  void* p = c.ws + c.off;
  c.off += bytes;
  return p;
}
static float* aalloc(Ctx& c, size_t n) { return (float*)aallocB(c, n * 4); }
static _Float16* aallocH(Ctx& c, size_t n) { return (_Float16*)aallocB(c, n * 2); }

static inline dim3 g1(long long n) { return dim3((unsigned)((n + TPB - 1) / TPB)); }

static void gemm_packed(Ctx& c, const _Float16* Ap, int Kp, const _Float16* Bp,
                        int N16, const float* bias, float* C_, int ldc,
                        int M, int N) {
  const int Mt = (M + 15) / 16;
  dim3 grid((Mt + WAVES_PER_BLOCK - 1) / WAVES_PER_BLOCK, N16 / 16);
  k_gemm_f16<<<grid, TPB, 0, c.st>>>(Ap, Kp, Bp, N16, bias, C_, ldc, M, N);
}

static void conv_raw(Ctx& c, int spec, const float* in, int H, int W, int stride,
                     float* out) {
  const Spec s = SPECS[spec];
  const int Ho = (H + stride - 1) / stride, Wo = (W + stride - 1) / stride;
  int pT = (Ho - 1) * stride + s.k - H; if (pT < 0) pT = 0;
  int pL = (Wo - 1) * stride + s.k - W; if (pL < 0) pL = 0;
  const int Ktot = s.k * s.k * s.cin;
  const int Kp = (Ktot + 31) & ~31;
  const int N16 = (s.cout + 15) & ~15;
  const int M = Ho * Wo;
  const size_t mark = c.off;
  // pack weights once per conv
  const long long btotal = (long long)(Kp / 32) * N16 * 32;
  _Float16* Bp = aallocH(c, (size_t)btotal);
  k_packB<<<g1(btotal), TPB, 0, c.st>>>(PW(c, spec), Ktot, s.cout, s.cout, 0,
                                        N16, btotal, Bp);
  // process output pixels in strips to bound im2col scratch
  const int strip = M < CONV_STRIP ? M : CONV_STRIP;
  _Float16* Ap = aallocH(c, (size_t)strip * Kp);
  for (int r0 = 0; r0 < M; r0 += strip) {
    const int rows = (M - r0 < strip) ? (M - r0) : strip;
    k_im2col<<<g1((long long)rows * Kp), TPB, 0, c.st>>>(
        in, H, W, s.cin, Wo, stride, s.k, s.k, pT / 2, pL / 2, r0,
        (long long)rows, Kp, Ktot, Ap);
    gemm_packed(c, Ap, Kp, Bp, N16, PB(c, spec), out + (size_t)r0 * s.cout,
                s.cout, rows, s.cout);
  }
  c.off = mark;
}

// generic f32 GEMM via pack + packed WMMA GEMM
static void gemm(Ctx& c, const float* A, int lda, const float* B, int ldb, int tB,
                 const float* bias, float* C_, int ldc, int M, int N, int K) {
  const size_t mark = c.off;
  const int Kp = (K + 31) & ~31;
  const int N16 = (N + 15) & ~15;
  _Float16* Ap = aallocH(c, (size_t)M * Kp);
  k_packA<<<g1((long long)M * Kp), TPB, 0, c.st>>>(A, lda, M, K, Kp, Ap);
  const long long btotal = (long long)(Kp / 32) * N16 * 32;
  _Float16* Bp = aallocH(c, (size_t)btotal);
  k_packB<<<g1(btotal), TPB, 0, c.st>>>(B, K, N, ldb, tB, N16, btotal, Bp);
  gemm_packed(c, Ap, Kp, Bp, N16, bias, C_, ldc, M, N);
  c.off = mark;
}

static float* gconv(Ctx& c, int spec, const float* in, int H, int W, int stride,
                    int act) {
  const Spec s = SPECS[spec];
  const int Ho = (H + stride - 1) / stride, Wo = (W + stride - 1) / stride;
  const bool gated = (act != ACT_NONE) && (s.cout != 3);
  const int Ce = gated ? s.cout / 2 : s.cout;
  float* outp = aalloc(c, (size_t)Ho * Wo * Ce);
  const size_t mark = c.off;
  if (gated) {
    float* raw = aalloc(c, (size_t)Ho * Wo * s.cout);
    conv_raw(c, spec, in, H, W, stride, raw);
    k_gate<<<g1((long long)Ho * Wo * Ce), TPB, 0, c.st>>>(
        raw, outp, (long long)Ho * Wo, Ce, act == ACT_RELU ? 1 : 0);
  } else {
    conv_raw(c, spec, in, H, W, stride, outp);
  }
  c.off = mark;
  return outp;
}

static float* gdconv(Ctx& c, int spec, const float* in, int H, int W) {
  const Spec s = SPECS[spec];
  const int H2 = 2 * H, W2 = 2 * W;
  const int Ce = s.cout / 2;
  float* outp = aalloc(c, (size_t)H2 * W2 * Ce);
  const size_t mark = c.off;
  float* up = aalloc(c, (size_t)H2 * W2 * s.cin);
  k_up2<<<g1((long long)H2 * W2 * s.cin), TPB, 0, c.st>>>(in, H, W, s.cin, up);
  float* raw = aalloc(c, (size_t)H2 * W2 * s.cout);
  conv_raw(c, spec, up, H2, W2, 1, raw);
  k_gate<<<g1((long long)H2 * W2 * Ce), TPB, 0, c.st>>>(raw, outp,
                                                        (long long)H2 * W2, Ce, 0);
  c.off = mark;
  return outp;
}

static float* concatN(Ctx& c, long long HW, const float** srcs, const int* cs,
                      int ns) {
  int Ct = 0; for (int i = 0; i < ns; ++i) Ct += cs[i];
  float* d = aalloc(c, (size_t)HW * Ct);
  int off = 0;
  for (int i = 0; i < ns; ++i) {
    k_copych<<<g1(HW * cs[i]), TPB, 0, c.st>>>(d, Ct, off, srcs[i], HW, cs[i]);
    off += cs[i];
  }
  return d;
}

static float* non_local(Ctx& c, const float* xs, const float* xd, int n, int pidx,
                        float* big) {
  const long long HW = (long long)n * n, HW4 = HW / 4;
  float* outp = aalloc(c, (size_t)HW * 48);
  const size_t mark = c.off;
  const float* Wth = (const float*)c.din[pidx + 0];
  const float* Wph = (const float*)c.din[pidx + 1];
  const float* Wg  = (const float*)c.din[pidx + 2];
  const float* Wo  = (const float*)c.din[pidx + 3];
  const float* sig = (const float*)c.din[pidx + 4];
  float* th = aalloc(c, (size_t)HW * 6);
  gemm(c, xs, 48, Wth, 6, 0, nullptr, th, 6, (int)HW, 6, 48);
  float* pt = aalloc(c, (size_t)HW * 6);
  gemm(c, xs, 48, Wph, 6, 0, nullptr, pt, 6, (int)HW, 6, 48);
  float* ph = aalloc(c, (size_t)HW4 * 6);
  k_maxpool2<<<g1(HW4 * 6), TPB, 0, c.st>>>(pt, n, n, 6, ph);
  float* gt = aalloc(c, (size_t)HW * 24);
  gemm(c, xd, 48, Wg, 24, 0, nullptr, gt, 24, (int)HW, 24, 48);
  float* gp = aalloc(c, (size_t)HW4 * 24);
  k_maxpool2<<<g1(HW4 * 24), TPB, 0, c.st>>>(gt, n, n, 24, gp);
  float* o = aalloc(c, (size_t)HW * 24);
  const long long CH = 4096;                     // attention row chunk
  for (long long r0 = 0; r0 < HW; r0 += CH) {
    const long long rows = (HW - r0 < CH) ? (HW - r0) : CH;
    gemm(c, th + r0 * 6, 6, ph, 6, 1, nullptr, big, (int)HW4,
         (int)rows, (int)HW4, 6);
    k_softmax_rows<<<dim3((unsigned)rows), TPB, 0, c.st>>>(big, nullptr, HW4, 1.f);
    gemm(c, big, (int)HW4, gp, 24, 0, nullptr, o + r0 * 24, 24,
         (int)rows, 24, (int)HW4);
  }
  float* o2 = aalloc(c, (size_t)HW * 48);
  gemm(c, o, 24, Wo, 48, 0, nullptr, o2, 48, (int)HW, 48, 24);
  k_addsig<<<g1(HW * 48), TPB, 0, c.st>>>(outp, xd, o2, sig, HW * 48);
  c.off = mark;
  return outp;
}

static float* ctx_attn(Ctx& c, const float* f, int S, const float* mask256,
                       float* big1, float* big2) {
  const int n = S / 2;
  const long long L = (long long)n * n;
  float* outp = aalloc(c, (size_t)S * S * 48);
  const size_t mark = c.off;
  float* fds = aalloc(c, (size_t)L * 48);
  k_subsample<<<g1(L * 48), TPB, 0, c.st>>>(f, S, 48, 2, n, n, fds);
  const int ms = 512 / S;                        // (256/S) * rate
  float* mds = aalloc(c, (size_t)L);
  k_subsample<<<g1(L), TPB, 0, c.st>>>(mask256, 256, 1, ms, n, n, mds);
  float* mmv = aalloc(c, (size_t)L);
  k_mm<<<g1(L), TPB, 0, c.st>>>(mds, n, mmv);
  float* invn = aalloc(c, (size_t)L);
  k_invnorm<<<g1(L), TPB, 0, c.st>>>(fds, n, 48, invn);
  // correlation: yi = (f patches) . (normalized b patches)^T via WMMA
  const int Kt = 9 * 48;
  const int Kp = (Kt + 31) & ~31;
  _Float16* Ap = aallocH(c, (size_t)L * Kp);
  k_patchA<<<g1(L * Kp), TPB, 0, c.st>>>(fds, n, 48, Kp, Ap);
  const long long btotal = (long long)(Kp / 32) * L * 32;
  _Float16* Bp = aallocH(c, (size_t)btotal);
  k_patchB<<<g1(btotal), TPB, 0, c.st>>>(fds, invn, n, 48, (int)L, btotal, Bp);
  gemm_packed(c, Ap, Kp, Bp, (int)L, nullptr, big1, (int)L, (int)L, (int)L);
  k_fuse1<<<g1(L * L), TPB, 0, c.st>>>(big1, big2, L, L);
  k_fuse2<<<g1(L * L), TPB, 0, c.st>>>(big2, big1, n);
  k_softmax_rows<<<dim3((unsigned)L), TPB, 0, c.st>>>(big1, mmv, L, 10.f);
  k_ca_deconv<<<dim3((unsigned)(S * S)), 64, 0, c.st>>>(big1, f, outp, n, 48);
  c.off = mark;
  return outp;
}

extern "C" void kernel_launch(void* const* d_in, const int* in_sizes, int n_in,
                              void* d_out, int out_size, void* d_ws, size_t ws_size,
                              hipStream_t stream) {
  Ctx c; c.din = d_in; c.ws = (char*)d_ws; c.cap = ws_size; c.off = 0; c.st = stream;
  const float* x0   = (const float*)d_in[0];
  const float* x    = (const float*)d_in[1];
  const float* mask = (const float*)d_in[2];
  const int NL64 = 3 + 2 * NSPEC;                // 109
  const int NL128 = NL64 + 5;                    // 114

  float* big1 = aalloc(c, (size_t)4096 * 4096);
  float* big2 = aalloc(c, (size_t)4096 * 4096);

  // ---- encoder (masked image branch) ----
  float* h5 = aalloc(c, (size_t)256 * 256 * 5);
  k_pack<<<g1(256 * 256), TPB, 0, c.st>>>(x, mask, h5, 256 * 256);
  float* t;
  t = gconv(c, S_conv1, h5, 256, 256, 1, ACT_ELU);          // 256,256,24
  t = gconv(c, S_conv1_ds, t, 256, 256, 2, ACT_ELU);        // 128,128,48
  t = gconv(c, S_conv2, t, 128, 128, 1, ACT_ELU);           // 128,128,48
  t = gconv(c, S_conv2_ds, t, 128, 128, 2, ACT_ELU);        // 64,64,96
  float* x64_e = gconv(c, S_conv3, t, 64, 64, 1, ACT_ELU);  // 64,64,96
  t = gconv(c, S_conv3_ds, x64_e, 64, 64, 2, ACT_ELU);      // 32,32,96
  float* x32_e = gconv(c, S_conv4, t, 32, 32, 1, ACT_ELU);  // 32,32,96
  t = gconv(c, S_conv4_ds, x32_e, 32, 32, 2, ACT_ELU);      // 16,16,96
  float* x16_e = gconv(c, S_conv5, t, 16, 16, 1, ACT_ELU);  // 16,16,96
  t = gconv(c, S_conv5_ds, x16_e, 16, 16, 2, ACT_ELU);      // 8,8,96
  float* x8_e = gconv(c, S_conv6, t, 8, 8, 1, ACT_ELU);     // 8,8,96
  float* x_8_8 = gconv(c, S_conv7, x8_e, 8, 8, 1, ACT_ELU); // 8,8,24

  // ---- encoder (original image branch) ----
  t = gconv(c, S_conv000_ds, x0, 256, 256, 2, ACT_ELU);     // 128,128,48
  float* x128 = gconv(c, S_conv001, t, 128, 128, 1, ACT_ELU);
  t = gconv(c, S_conv001_ds, x128, 128, 128, 2, ACT_ELU);   // 64,64,48
  float* x64s = gconv(c, S_conv002, t, 64, 64, 1, ACT_ELU); // 64,64,48
  t = gconv(c, S_conv002_ds, x64s, 64, 64, 2, ACT_ELU);     // 32,32,48
  float* x32s = gconv(c, S_conv003, t, 32, 32, 1, ACT_ELU); // 32,32,48
  (void)x32s;

  // ---- decoder ----
  { const float* s2[2] = { x8_e, x_8_8 }; const int cc[2] = { 96, 24 };
    t = concatN(c, 8 * 8, s2, cc, 2); }                      // x8_d 120
  float* x8_up_16 = gdconv(c, S_conv7_up, t, 8, 8);          // 16,16,24
  { const float* s2[2] = { x16_e, x8_up_16 }; const int cc[2] = { 96, 24 };
    t = concatN(c, 16 * 16, s2, cc, 2); }                    // x16_d 120
  float* x16_up_32 = gdconv(c, S_conv8_up, t, 16, 16);       // 32,32,24
  float* x16_32 = gdconv(c, S_conv81_up, x8_up_16, 16, 16);  // 32,32,24
  float* x32_d;
  { const float* s3[3] = { x32_e, x16_up_32, x16_32 }; const int cc[3] = { 96, 24, 24 };
    x32_d = concatN(c, 32 * 32, s3, cc, 3); }                // 144
  x32_d = gconv(c, S_conv_self32_1, x32_d, 32, 32, 1, ACT_ELU);      // 48
  float* x32att = gconv(c, S_pmconv32, x32_d, 32, 32, 1, ACT_RELU);  // 48
  float* att32 = ctx_attn(c, x32att, 32, mask, big1, big2);          // 32,32,48
  att32 = gconv(c, S_conv9, att32, 32, 32, 1, ACT_ELU);              // 48
  float* x32_d_64 = gdconv(c, S_conv9_up, att32, 32, 32);            // 64,64,48
  x32_d_64 = gconv(c, S_conv10, x32_d_64, 64, 64, 1, ACT_ELU);       // 48
  float* x32_up_64 = gdconv(c, S_conv10_up, x32_d, 32, 32);          // 64,64,24
  float* x32_64 = gdconv(c, S_conv101_up, x16_up_32, 32, 32);        // 64,64,24
  float* x64_d;
  { const float* s4[4] = { x64_e, x32_d_64, x32_up_64, x32_64 };
    const int cc[4] = { 96, 48, 24, 24 };
    x64_d = concatN(c, 64 * 64, s4, cc, 4); }                        // 192
  x64_d = gconv(c, S_conv10_1, x64_d, 64, 64, 1, ACT_ELU);           // 48
  float* att64nl = non_local(c, x64s, x64_d, 64, NL64, big1);
  float* x64_att;
  { const float* s2[2] = { x64_d, att64nl }; const int cc[2] = { 48, 48 };
    x64_att = concatN(c, 64 * 64, s2, cc, 2); }                      // 96
  x64_att = gconv(c, S_conv10_2, x64_att, 64, 64, 1, ACT_ELU);       // 48
  float* x64att_d = gconv(c, S_pmconv64_d, x64_d, 64, 64, 1, ACT_RELU);
  float* att64_d = ctx_attn(c, x64att_d, 64, mask, big1, big2);
  float* x64att = gconv(c, S_pmconv64, x64_att, 64, 64, 1, ACT_RELU);
  float* att64 = ctx_attn(c, x64att, 64, mask, big1, big2);
  att64_d = gconv(c, S_conv11_d, att64_d, 64, 64, 1, ACT_ELU);       // 48
  att64 = gconv(c, S_conv11, att64, 64, 64, 1, ACT_ELU);             // 48
  float* x64_d_128_d = gdconv(c, S_conv11_up_d, att64_d, 64, 64);    // 128,128,48
  float* x64_d_128 = gdconv(c, S_conv11_up, att64, 64, 64);          // 128,128,48
  x64_d_128 = gconv(c, S_conv12, x64_d_128, 128, 128, 1, ACT_ELU);   // 48
  float* x64_up_128 = gdconv(c, S_conv12_up, x64_d, 64, 64);         // 128,128,24
  float* x64_128 = gdconv(c, S_conv121_up, x32_up_64, 64, 64);       // 128,128,24
  float* x128_d;
  { const float* s4[4] = { x64_d_128, x64_up_128, x64_128, x64_d_128_d };
    const int cc[4] = { 48, 24, 24, 48 };
    x128_d = concatN(c, 128 * 128, s4, cc, 4); }                     // 144
  x128_d = gconv(c, S_conv12_1, x128_d, 128, 128, 1, ACT_ELU);       // 48
  float* att128nl = non_local(c, x128, x128_d, 128, NL128, big1);
  float* x128_att;
  { const float* s2[2] = { x128_d, att128nl }; const int cc[2] = { 48, 48 };
    x128_att = concatN(c, 128 * 128, s2, cc, 2); }                   // 96
  x128_att = gconv(c, S_conv12_2, x128_att, 128, 128, 1, ACT_ELU);   // 48
  float* x128att_d = gconv(c, S_pmconv128, x128_d, 128, 128, 1, ACT_RELU);
  float* att128_d = ctx_attn(c, x128att_d, 128, mask, big1, big2);
  float* x128att = gconv(c, S_pmconv128_d, x128_att, 128, 128, 1, ACT_RELU);
  float* att128 = ctx_attn(c, x128att, 128, mask, big1, big2);
  att128_d = gconv(c, S_conv13_d, att128_d, 128, 128, 1, ACT_ELU);   // 48
  att128 = gconv(c, S_conv13, att128, 128, 128, 1, ACT_ELU);         // 48
  float* x128_d_256_d = gdconv(c, S_conv13_up_d, att128_d, 128, 128);// 256,256,48
  float* x128_d_256 = gdconv(c, S_conv13_up, att128, 128, 128);      // 256,256,48
  x128_d_256 = gconv(c, S_conv14, x128_d_256, 256, 256, 1, ACT_ELU); // 24
  float* x128_up_256 = gdconv(c, S_conv14_up, x128_d, 128, 128);     // 256,256,24
  float* x128_256 = gdconv(c, S_conv15_up, x64_up_128, 128, 128);    // 256,256,24
  float* x256_d;
  { const float* s4[4] = { x128_d_256, x128_256, x128_up_256, x128_d_256_d };
    const int cc[4] = { 24, 24, 24, 48 };
    x256_d = concatN(c, 256 * 256, s4, cc, 4); }                     // 120
  x256_d = gconv(c, S_conv15, x256_d, 256, 256, 1, ACT_ELU);         // 24
  float* h16 = gconv(c, S_conv16, x256_d, 256, 256, 1, ACT_ELU);     // 12
  float* h17 = gconv(c, S_conv17, h16, 256, 256, 1, ACT_NONE);       // 3
  k_tanh<<<g1((long long)256 * 256 * 3), TPB, 0, c.st>>>(h17, (float*)d_out,
                                                         (long long)256 * 256 * 3);
}
